// MambaBlock_43533788512997
// MI455X (gfx1250) — matrix operand order repaired and checked by reading, not compiled, not run on hardware
//
#include <hip/hip_runtime.h>
#include <cmath>

// ---------------------------------------------------------------------------
// Mamba block sizes (compile-time constants from the reference)
// ---------------------------------------------------------------------------
#define D_MODEL  1024
#define D_STATE  16
#define D_CONV   4
#define D_INNER  2048          // EXPAND * D_MODEL
#define DT_RANK  64            // ceil(1024/16)
#define B_SZ     2
#define L_SEQ    2048
#define M_ROWS   (B_SZ * L_SEQ)   // 4096 flattened tokens
#define XDBL_W   96               // DT_RANK + 2*D_STATE
#define XDBL_PAD 128              // padded to a WMMA-friendly width

typedef __attribute__((ext_vector_type(16))) __bf16 v16bf;
typedef __attribute__((ext_vector_type(8)))  __bf16 v8bf;
typedef __attribute__((ext_vector_type(8)))  float  v8f;

__device__ __forceinline__ float silu_f(float x)     { return x / (1.0f + __expf(-x)); }
__device__ __forceinline__ float softplus_f(float x) { return (x > 20.0f) ? x : log1pf(__expf(x)); }

// Convert 8 consecutive fp32 to a packed v8bf (compiler vectorizes the loads,
// emits cvt + pack, and lets us use 16-byte ds_store_b128 for LDS staging).
__device__ __forceinline__ v8bf cvt8(const float* __restrict__ s)
{
    v8bf v;
#pragma unroll
    for (int i = 0; i < 8; ++i) v[i] = (__bf16)s[i];
    return v;
}

// ---------------------------------------------------------------------------
// Tiled bf16 WMMA GEMM:  C[M,N] = A[M,K] * B[K,N]   (fp32 in/out, bf16 MACs,
// f32 accumulate).  Block tile 256x128, BK=32, 256 threads = 8 waves laid out
// 4(M) x 2(N); each wave owns a 64x64 sub-tile = 4x4 16x16x32 accumulators
// (16 v_wmma per K-step, 8 ds_load_b128-pair fragment loads).
// Requires: M multiple of 256, N multiple of 128, K multiple of 32.
// ---------------------------------------------------------------------------
__global__ __launch_bounds__(256)
void gemm_bf16_wmma_kernel(const float* __restrict__ A,
                           const float* __restrict__ Bg,
                           float* __restrict__ C,
                           int N, int K, int lda, int ldb, int ldc)
{
    __shared__ alignas(16) __bf16 As[256 * 32];  // [row][k], k contiguous
    __shared__ alignas(16) __bf16 Bs[128 * 32];  // [col][k], k contiguous (B^T)

    const int tid   = threadIdx.x;
    const int lane  = tid & 31;
    const int wave  = tid >> 5;
    const int wm    = wave >> 1;     // 0..3 -> 64-row strip
    const int wn    = wave & 1;      // 0..1 -> 64-col strip
    const int lrow  = lane & 15;
    const int khalf = lane >> 4;     // which K half-group this lane holds

    const int blockRow = blockIdx.y * 256;
    const int blockCol = blockIdx.x * 128;

    __builtin_assume(K >= 32);

    v8f acc[4][4];
#pragma unroll
    for (int mi = 0; mi < 4; ++mi)
#pragma unroll
        for (int ni = 0; ni < 4; ++ni)
#pragma unroll
            for (int r = 0; r < 8; ++r) acc[mi][ni][r] = 0.0f;

    for (int k0 = 0; k0 < K; k0 += 32) {
        // --- stage A tile (256x32): one row per thread, 32 contiguous fp32
        //     (2x global_load_b128x2) -> 4x ds_store_b128 of bf16. ---
        {
            const float* src = A + (size_t)(blockRow + tid) * lda + k0;
            __bf16* dst = As + tid * 32;
#pragma unroll
            for (int q = 0; q < 4; ++q)
                *(v8bf*)(dst + q * 8) = cvt8(src + q * 8);
        }
        // --- stage B tile (32x128) transposed into [n][k]: one column half
        //     per thread.  Lanes with consecutive n read consecutive addrs
        //     per k-row -> coalesced; LDS writes are 2x ds_store_b128. ---
        {
            const int n  = tid & 127;
            const int kg = tid >> 7;          // 0/1 -> k 0..15 / 16..31
            const float* src = Bg + (size_t)(k0 + kg * 16) * ldb + blockCol + n;
            float tmp[16];
#pragma unroll
            for (int j = 0; j < 16; ++j) tmp[j] = src[(size_t)j * ldb];
            __bf16* dst = Bs + n * 32 + kg * 16;
            *(v8bf*)(dst)     = cvt8(tmp);
            *(v8bf*)(dst + 8) = cvt8(tmp + 8);
        }
        __syncthreads();

        // --- fragments: contiguous k order, same in A and B ---
        // A 16x32 bf16: elems 0-15 -> K = khalf*16 + 0..15
        v16bf afr[4];
#pragma unroll
        for (int mi = 0; mi < 4; ++mi) {
            const __bf16* base = As + (wm * 64 + mi * 16 + lrow) * 32 + khalf * 16;
            v8bf lo = *(const v8bf*)(base);
            v8bf hi = *(const v8bf*)(base + 8);
            afr[mi] = __builtin_shufflevector(lo, hi,
                        0,1,2,3,4,5,6,7,8,9,10,11,12,13,14,15);
        }
        // B 32x16 bf16: elems 0-15 -> K = khalf*16 + 0..15
        v16bf bfr[4];
#pragma unroll
        for (int ni = 0; ni < 4; ++ni) {
            const __bf16* base = Bs + (wn * 64 + ni * 16 + lrow) * 32 + khalf * 16;
            v8bf lo = *(const v8bf*)(base);
            v8bf hi = *(const v8bf*)(base + 8);
            bfr[ni] = __builtin_shufflevector(lo, hi,
                        0,1,2,3,4,5,6,7,8,9,10,11,12,13,14,15);
        }

#pragma unroll
        for (int mi = 0; mi < 4; ++mi)
#pragma unroll
            for (int ni = 0; ni < 4; ++ni)
                acc[mi][ni] = __builtin_amdgcn_wmma_f32_16x16x32_bf16(
                    /*neg_a=*/false, afr[mi], /*neg_b=*/false, bfr[ni],
                    /*c_mod=*/(short)0, acc[mi][ni],
                    /*reuse_a=*/false, /*reuse_b=*/false);
        __syncthreads();
    }

    // --- writeback: C/D layout: vgpr r, lane -> M = r + 8*khalf, N = lrow ---
#pragma unroll
    for (int mi = 0; mi < 4; ++mi) {
#pragma unroll
        for (int ni = 0; ni < 4; ++ni) {
            const int col  = blockCol + wn * 64 + ni * 16 + lrow;
            const int row0 = blockRow + wm * 64 + mi * 16 + khalf * 8;
#pragma unroll
            for (int r = 0; r < 8; ++r)
                C[(size_t)(row0 + r) * ldc + col] = acc[mi][ni][r];
        }
    }
}

// ---------------------------------------------------------------------------
// Zero-pad W_x (2048 x 96) -> (2048 x 128)
// ---------------------------------------------------------------------------
__global__ void pad_wx_kernel(const float* __restrict__ Wx, float* __restrict__ Wp)
{
    int i = blockIdx.x * blockDim.x + threadIdx.x;
    if (i >= D_INNER * XDBL_PAD) return;
    int r = i >> 7, c = i & (XDBL_PAD - 1);
    Wp[i] = (c < XDBL_W) ? Wx[r * XDBL_W + c] : 0.0f;
}

// ---------------------------------------------------------------------------
// Depthwise causal conv (k=4) + bias + SiLU.  xs lives in xr cols [0,2048).
// ---------------------------------------------------------------------------
__global__ void conv_silu_kernel(const float* __restrict__ xr,
                                 const float* __restrict__ cw,
                                 const float* __restrict__ cb,
                                 float* __restrict__ xc)
{
    int i = blockIdx.x * blockDim.x + threadIdx.x;
    if (i >= M_ROWS * D_INNER) return;
    const int m = i / D_INNER;
    const int d = i - m * D_INNER;
    const int l = m & (L_SEQ - 1);
    float acc = cb[d];
#pragma unroll
    for (int k = 0; k < D_CONV; ++k) {
        const int li = l + k - (D_CONV - 1);
        if (li >= 0)
            acc = fmaf(cw[d * D_CONV + k],
                       xr[(size_t)(m + k - (D_CONV - 1)) * (2 * D_INNER) + d], acc);
    }
    xc[i] = silu_f(acc);
}

// ---------------------------------------------------------------------------
// In-place softplus(delta_raw + b_dt); delta stored strided (ld = 4096) in xr.
// ---------------------------------------------------------------------------
__global__ void softplus_bias_kernel(float* __restrict__ deltaS,
                                     const float* __restrict__ b_dt)
{
    int i = blockIdx.x * blockDim.x + threadIdx.x;
    if (i >= M_ROWS * D_INNER) return;
    const int m = i / D_INNER;
    const int d = i - m * D_INNER;
    float* p = deltaS + (size_t)m * (2 * D_INNER) + d;
    *p = softplus_f(*p + b_dt[d]);
}

// ---------------------------------------------------------------------------
// Selective scan: one thread per (b,d) channel, h[16] in registers,
// sequential over L.  B/C rows are uniform per block -> broadcast loads.
// ---------------------------------------------------------------------------
__global__ __launch_bounds__(256)
void scan_kernel(const float* __restrict__ deltaS,  // ld = 4096
                 const float* __restrict__ xc,      // ld = 2048
                 const float* __restrict__ xdbl,    // ld = 128, B at +64, C at +80
                 const float* __restrict__ A_log,
                 float* __restrict__ ys)
{
    const int d = blockIdx.x * blockDim.x + threadIdx.x;
    const int b = blockIdx.y;

    float Ad[D_STATE];
#pragma unroll
    for (int n = 0; n < D_STATE; ++n) Ad[n] = -__expf(A_log[d * D_STATE + n]);

    float h[D_STATE];
#pragma unroll
    for (int n = 0; n < D_STATE; ++n) h[n] = 0.0f;

    for (int l = 0; l < L_SEQ; ++l) {
        const int m = b * L_SEQ + l;
        const float dv = deltaS[(size_t)m * (2 * D_INNER) + d];
        const float xv = xc[(size_t)m * D_INNER + d];
        const float dxb = dv * xv;
        const float* bc = xdbl + (size_t)m * XDBL_PAD + DT_RANK;
        float y = 0.0f;
#pragma unroll
        for (int n = 0; n < D_STATE; ++n) {
            const float dA = __expf(dv * Ad[n]);
            h[n] = fmaf(dA, h[n], dxb * bc[n]);
            y    = fmaf(h[n], bc[D_STATE + n], y);
        }
        ys[(size_t)m * D_INNER + d] = y;
    }
}

// ---------------------------------------------------------------------------
// y = (ys + xc*D) * silu(res);  res lives in xr cols [2048,4096).  In place.
// ---------------------------------------------------------------------------
__global__ void ymod_kernel(float* __restrict__ ys,
                            const float* __restrict__ xc,
                            const float* __restrict__ Dp,
                            const float* __restrict__ xr)
{
    int i = blockIdx.x * blockDim.x + threadIdx.x;
    if (i >= M_ROWS * D_INNER) return;
    const int m = i / D_INNER;
    const int d = i - m * D_INNER;
    const float res = xr[(size_t)m * (2 * D_INNER) + D_INNER + d];
    ys[i] = fmaf(xc[i], Dp[d], ys[i]) * silu_f(res);
}

// ---------------------------------------------------------------------------
// Launch
// ---------------------------------------------------------------------------
extern "C" void kernel_launch(void* const* d_in, const int* in_sizes, int n_in,
                              void* d_out, int out_size, void* d_ws, size_t ws_size,
                              hipStream_t stream)
{
    (void)in_sizes; (void)n_in; (void)out_size; (void)ws_size;

    const float* x      = (const float*)d_in[0];
    const float* W_in   = (const float*)d_in[1];
    const float* conv_w = (const float*)d_in[2];
    const float* conv_b = (const float*)d_in[3];
    const float* W_x    = (const float*)d_in[4];
    const float* W_dt   = (const float*)d_in[5];
    const float* b_dt   = (const float*)d_in[6];
    const float* A_log  = (const float*)d_in[7];
    const float* D_par  = (const float*)d_in[8];
    const float* W_out  = (const float*)d_in[9];
    float* out = (float*)d_out;

    // Workspace carve-up (floats): ~131 MB total
    float* ws     = (float*)d_ws;
    float* xr     = ws;                                   // 4096 x 4096 (xs | res)
    float* xc     = xr   + (size_t)M_ROWS * 2 * D_INNER;  // 4096 x 2048
    float* xdbl   = xc   + (size_t)M_ROWS * D_INNER;      // 4096 x 128
    float* Wx_pad = xdbl + (size_t)M_ROWS * XDBL_PAD;     // 2048 x 128
    float* ys     = Wx_pad + (size_t)D_INNER * XDBL_PAD;  // 4096 x 2048
    float* deltaS = xr;  // delta overwrites dead xs half of xr (ld = 4096)

    const int EW = 256;
    const int nElem = M_ROWS * D_INNER;

    // 0. pad W_x to 128 cols
    pad_wx_kernel<<<(D_INNER * XDBL_PAD + EW - 1) / EW, EW, 0, stream>>>(W_x, Wx_pad);

    // 1. xr = x @ W_in        (4096x1024)(1024x4096)
    gemm_bf16_wmma_kernel<<<dim3((2 * D_INNER) / 128, M_ROWS / 256), 256, 0, stream>>>(
        x, W_in, xr, 2 * D_INNER, D_MODEL, D_MODEL, 2 * D_INNER, 2 * D_INNER);

    // 2. xc = silu(depthwise_conv(xs) + b)
    conv_silu_kernel<<<(nElem + EW - 1) / EW, EW, 0, stream>>>(xr, conv_w, conv_b, xc);

    // 3. x_dbl = xc @ W_x_pad (4096x2048)(2048x128)
    gemm_bf16_wmma_kernel<<<dim3(XDBL_PAD / 128, M_ROWS / 256), 256, 0, stream>>>(
        xc, Wx_pad, xdbl, XDBL_PAD, D_INNER, D_INNER, XDBL_PAD, XDBL_PAD);

    // 4. delta_raw = x_dbl[:, :64] @ W_dt  -> strided into xr cols [0,2048)
    gemm_bf16_wmma_kernel<<<dim3(D_INNER / 128, M_ROWS / 256), 256, 0, stream>>>(
        xdbl, W_dt, deltaS, D_INNER, DT_RANK, XDBL_PAD, D_INNER, 2 * D_INNER);

    // 5. delta = softplus(delta_raw + b_dt)   (in place, strided)
    softplus_bias_kernel<<<(nElem + EW - 1) / EW, EW, 0, stream>>>(deltaS, b_dt);

    // 6. selective scan -> ys
    scan_kernel<<<dim3(D_INNER / 256, B_SZ), 256, 0, stream>>>(
        deltaS, xc, xdbl, A_log, ys);

    // 7. ys = (ys + xc*D) * silu(res)
    ymod_kernel<<<(nElem + EW - 1) / EW, EW, 0, stream>>>(ys, xc, D_par, xr);

    // 8. out = ys @ W_out     (4096x2048)(2048x1024)
    gemm_bf16_wmma_kernel<<<dim3(D_MODEL / 128, M_ROWS / 256), 256, 0, stream>>>(
        ys, W_out, out, D_MODEL, D_INNER, D_INNER, D_MODEL, D_MODEL);
}
